// DeformConv_19473381720651
// MI455X (gfx1250) — matrix operand order repaired and checked by reading, not compiled, not run on hardware
//
#include <hip/hip_runtime.h>

// ---------------------------------------------------------------------------
// Deformable Conv v1 on MI455X (gfx1250, wave32).
//   K1: offset conv (fp32 VALU, 2.7 GFLOP)
//   K2: weight reorder+convert fp32 -> bf16, layout W[cout][kk*256+cin]
//   K3: fused bilinear-sample + implicit-im2col GEMM using
//       v_wmma_f32_16x16x32_bf16 (38.7 GFLOP, compute-bound; HBM ~75MB).
//       Block tile M=256 x N=128, 8 waves, 64x64 per wave = 16 WMMA/k-step.
//       A tile staged via global_load_async_to_lds_b128 (ASYNCcnt DMA path).
// ---------------------------------------------------------------------------

typedef __attribute__((ext_vector_type(16))) __bf16 v16bf;
typedef __attribute__((ext_vector_type(8)))  __bf16 v8bf;
typedef __attribute__((ext_vector_type(8)))  float  v8f;
typedef int v4i_ __attribute__((vector_size(16)));   // matches builtin param type

#define BATCH   8
#define CIN     256
#define COUT    256
#define KDIM    2304      // CIN * 9
#define NTOT    32768     // BATCH*64*64
#define TILE_N  128
#define NKSTEPS 72        // KDIM / 32
#define AP      40        // A LDS pitch (bf16 elems): 80B, 16B aligned
#define BP      40        // B LDS pitch

// ------------------------- Kernel 1: offset conv ---------------------------
__global__ __launch_bounds__(256)
void dcn_offs_conv(const float* __restrict__ x,
                   const float* __restrict__ w_off,
                   const float* __restrict__ b_off,
                   float* __restrict__ offs)
{
    const int n  = blockIdx.x * 256 + threadIdx.x;   // 0..32767
    const int b  = n >> 12;
    const int hw = n & 4095;
    const int h  = hw >> 6;
    const int w  = hw & 63;

    float acc[18];
#pragma unroll
    for (int oc = 0; oc < 18; ++oc) acc[oc] = b_off[oc];

    const float* xb = x + ((size_t)b << 20);         // b * 256*4096
    for (int cin = 0; cin < CIN; ++cin) {
        const float* xp = xb + ((size_t)cin << 12);
        const float* wp = w_off + (size_t)cin * 9;   // stride per oc = 2304
        float v[9];
#pragma unroll
        for (int t = 0; t < 9; ++t) {
            const int ky = t / 3, kx = t - 3 * (t / 3);
            const int yy = h - 1 + ky;
            const int xx = w - 1 + kx;
            const bool ok = (yy >= 0) & (yy < 64) & (xx >= 0) & (xx < 64);
            v[t] = ok ? xp[(yy << 6) + xx] : 0.0f;
        }
#pragma unroll
        for (int oc = 0; oc < 18; ++oc) {
            float a = acc[oc];
#pragma unroll
            for (int t = 0; t < 9; ++t)
                a = fmaf(v[t], wp[(size_t)oc * KDIM + t], a);
            acc[oc] = a;
        }
    }
#pragma unroll
    for (int oc = 0; oc < 18; ++oc)
        offs[(((size_t)b * 18 + oc) << 12) + hw] = acc[oc];
}

// ----------------- Kernel 2: weight convert / reorder to bf16 --------------
__global__ __launch_bounds__(256)
void dcn_wcvt(const float* __restrict__ w_dcn, unsigned short* __restrict__ Wb_u)
{
    __bf16* Wb = (__bf16*)Wb_u;
    const int t = blockIdx.x * 256 + threadIdx.x;
    if (t >= COUT * KDIM) return;
    const int cout = t / KDIM;
    const int r    = t - cout * KDIM;
    const int kk   = r >> 8;
    const int cin  = r & 255;
    const float v = w_dcn[((size_t)cout * CIN + cin) * 9 + kk];
    Wb[(size_t)cout * KDIM + (kk << 8) + cin] = (__bf16)v;
}

// --------- Kernel 3: fused sample + implicit-im2col WMMA GEMM --------------
__global__ __launch_bounds__(256)
void dcn_gemm(const float* __restrict__ x,
              const unsigned short* __restrict__ Wb_u,
              const float* __restrict__ offs,
              const float* __restrict__ b_dcn,
              float* __restrict__ out)
{
    const __bf16* Wb = (const __bf16*)Wb_u;

    __shared__ __attribute__((aligned(16))) __bf16 Atile[COUT * AP];
    __shared__ __attribute__((aligned(16))) __bf16 Btile[TILE_N * BP];
    __shared__ float Pw[TILE_N][4];   // bilinear weights (validity-folded)
    __shared__ int   Po[TILE_N][4];   // clamped element offsets in x-plane

    const int tid    = threadIdx.x;
    const int lane   = tid & 31;
    const int wave   = tid >> 5;
    const int wave_m = wave & 3;      // 4 M-blocks of 64
    const int wave_n = wave >> 2;     // 2 N-blocks of 64
    const int n0     = blockIdx.x * TILE_N;
    const int b      = n0 >> 12;      // TILE_N | 4096 -> b constant per block

    v8f acc[4][4] = {};

    for (int kt = 0; kt < NKSTEPS; ++kt) {
        __syncthreads();  // previous iteration's WMMAs done before overwrite

        // --- per-tap bilinear parameters (every 8 K-steps) ---
        if ((kt & 7) == 0) {
            const int kk = kt >> 3;
            if (tid < TILE_N) {
                const int n  = n0 + tid;
                const int hw = n & 4095;
                const int h  = hw >> 6, w = hw & 63;
                const float dy = offs[(((size_t)b * 18 + 2 * kk) << 12) + hw];
                const float dx = offs[(((size_t)b * 18 + 2 * kk + 1) << 12) + hw];
                const int ky = kk / 3;
                const int kx = kk - 3 * ky;
                const float py = (float)(h - 1 + ky) + dy;
                const float px = (float)(w - 1 + kx) + dx;
                const float fy0 = floorf(py), fx0 = floorf(px);
                const int iy0 = (int)fy0, ix0 = (int)fx0;
                const int iy1 = iy0 + 1,  ix1 = ix0 + 1;
                const float fy = py - fy0, fx = px - fx0;
                const float vy0 = (iy0 >= 0 && iy0 < 64) ? 1.f : 0.f;
                const float vy1 = (iy1 >= 0 && iy1 < 64) ? 1.f : 0.f;
                const float vx0 = (ix0 >= 0 && ix0 < 64) ? 1.f : 0.f;
                const float vx1 = (ix1 >= 0 && ix1 < 64) ? 1.f : 0.f;
                Pw[tid][0] = (1.f - fy) * (1.f - fx) * vy0 * vx0;
                Pw[tid][1] = (1.f - fy) * fx         * vy0 * vx1;
                Pw[tid][2] = fy         * (1.f - fx) * vy1 * vx0;
                Pw[tid][3] = fy         * fx         * vy1 * vx1;
                const int cy0 = iy0 < 0 ? 0 : (iy0 > 63 ? 63 : iy0);
                const int cy1 = iy1 < 0 ? 0 : (iy1 > 63 ? 63 : iy1);
                const int cx0 = ix0 < 0 ? 0 : (ix0 > 63 ? 63 : ix0);
                const int cx1 = ix1 < 0 ? 0 : (ix1 > 63 ? 63 : ix1);
                Po[tid][0] = (cy0 << 6) + cx0;
                Po[tid][1] = (cy0 << 6) + cx1;
                Po[tid][2] = (cy1 << 6) + cx0;
                Po[tid][3] = (cy1 << 6) + cx1;
            }
            __syncthreads();
        }

        // --- stage A tile: 256 rows x 32 K of bf16 weights (async DMA) ---
        {
            const uint4* g = (const uint4*)(Wb + (size_t)tid * KDIM + kt * 32);
            uint4* s = (uint4*)(Atile + tid * AP);
#if __has_builtin(__builtin_amdgcn_global_load_async_to_lds_b128)
#pragma unroll
            for (int c = 0; c < 4; ++c)
                __builtin_amdgcn_global_load_async_to_lds_b128(
                    (__attribute__((address_space(1))) v4i_*)(g + c),
                    (__attribute__((address_space(3))) v4i_*)(s + c),
                    /*offset=*/0, /*cpol=*/0);
#else
            uint4 r0 = g[0], r1 = g[1], r2 = g[2], r3 = g[3];
            s[0] = r0; s[1] = r1; s[2] = r2; s[3] = r3;
#endif
            if (kt + 1 < NKSTEPS) __builtin_prefetch((const void*)(g + 4));
        }

        // --- build B tile: 128 positions x 32 channels, bilinear sampled ---
        {
            const int cj   = tid & 31;          // channel lane within chunk
            const int grp  = tid >> 5;          // 8 groups of 16 positions
            const int cin0 = (kt & 7) << 5;
            const float* xb = x + ((size_t)((b << 8) + cin0 + cj) << 12);
#pragma unroll
            for (int i = 0; i < 16; ++i) {
                const int nl = (grp << 4) + i;
                const float w0 = Pw[nl][0], w1 = Pw[nl][1];
                const float w2 = Pw[nl][2], w3 = Pw[nl][3];
                const int o0 = Po[nl][0], o1 = Po[nl][1];
                const int o2 = Po[nl][2], o3 = Po[nl][3];
                float s = w0 * xb[o0];
                s = fmaf(w1, xb[o1], s);
                s = fmaf(w2, xb[o2], s);
                s = fmaf(w3, xb[o3], s);
                Btile[nl * BP + cj] = (__bf16)s;
            }
        }

#if __has_builtin(__builtin_amdgcn_global_load_async_to_lds_b128)
#if __has_builtin(__builtin_amdgcn_s_wait_asynccnt)
        __builtin_amdgcn_s_wait_asynccnt(0);
#else
        asm volatile("s_wait_asynccnt 0x0" ::: "memory");
#endif
#endif
        __syncthreads();

        // --- fragment loads (ISA 7.12.2 layouts) + 16 WMMAs per wave ---
        v16bf afr[4];
        const int arow  = lane & 15;
        const int akoff = (lane < 16) ? 0 : 8;
#pragma unroll
        for (int mi = 0; mi < 4; ++mi) {
            const int row = wave_m * 64 + mi * 16 + arow;
            const __bf16* p = Atile + row * AP + akoff;
            v8bf lo = *(const v8bf*)p;          // K = akoff .. akoff+7
            v8bf hi = *(const v8bf*)(p + 16);   // K = akoff+16 .. akoff+23
#pragma unroll
            for (int i = 0; i < 8; ++i) { afr[mi][i] = lo[i]; afr[mi][8 + i] = hi[i]; }
        }
        const int bkoff = (lane < 16) ? 0 : 8;
#pragma unroll
        for (int ni = 0; ni < 4; ++ni) {
            const int nn = wave_n * 64 + ni * 16 + (lane & 15);
            const __bf16* p = Btile + nn * BP + bkoff;
            v8bf lo = *(const v8bf*)p;          // K = bkoff .. bkoff+7
            v8bf hi = *(const v8bf*)(p + 16);   // K = bkoff+16 .. bkoff+23
            v16bf bfr;
#pragma unroll
            for (int i = 0; i < 8; ++i) { bfr[i] = lo[i]; bfr[8 + i] = hi[i]; }
#pragma unroll
            for (int mi = 0; mi < 4; ++mi)
                acc[mi][ni] = __builtin_amdgcn_wmma_f32_16x16x32_bf16(
                    false, afr[mi], false, bfr,
                    (short)0, acc[mi][ni], false, false);
        }
    }

    // --- epilogue: bias + store (C/D layout: VGPR r -> M = r + 8*(lane>=16)) ---
    const int mhalf = (lane >> 4) << 3;
    const int ncol0 = n0 + wave_n * 64 + (lane & 15);
#pragma unroll
    for (int mi = 0; mi < 4; ++mi) {
#pragma unroll
        for (int ni = 0; ni < 4; ++ni) {
            const int hw = (ncol0 + ni * 16) & 4095;
#pragma unroll
            for (int r = 0; r < 8; ++r) {
                const int cout = wave_m * 64 + mi * 16 + mhalf + r;
                out[(((size_t)b * COUT + cout) << 12) + hw] =
                    acc[mi][ni][r] + b_dcn[cout];
            }
        }
    }
}

// --------------------------------- launch ----------------------------------
extern "C" void kernel_launch(void* const* d_in, const int* in_sizes, int n_in,
                              void* d_out, int out_size, void* d_ws, size_t ws_size,
                              hipStream_t stream)
{
    const float* x     = (const float*)d_in[0];   // [8,256,64,64]
    const float* w_off = (const float*)d_in[1];   // [18,256,3,3]
    const float* b_off = (const float*)d_in[2];   // [18]
    const float* w_dcn = (const float*)d_in[3];   // [256,256,3,3]
    const float* b_dcn = (const float*)d_in[4];   // [256]
    float* out = (float*)d_out;                   // [8,256,64,64]

    // workspace: offsets (fp32, 9.4MB) then reordered bf16 weights (1.2MB)
    float* offs = (float*)d_ws;
    unsigned short* Wb = (unsigned short*)((char*)d_ws + (size_t)2359296 * 4);

    dcn_offs_conv<<<NTOT / 256, 256, 0, stream>>>(x, w_off, b_off, offs);
    dcn_wcvt<<<(COUT * KDIM) / 256, 256, 0, stream>>>(w_dcn, Wb);
    dcn_gemm<<<NTOT / TILE_N, 256, 0, stream>>>(x, Wb, offs, b_dcn, out);
}
